// S_MH_MLP_48773648614361
// MI455X (gfx1250) — compile-verified, matrix operand order checked by reading, not run on hardware
//
#include <hip/hip_runtime.h>

typedef __attribute__((ext_vector_type(16))) __bf16 v16bf;
typedef __attribute__((ext_vector_type(8)))  float  v8f;

#define B_   8
#define S_   2048
#define D_   1024
#define H_   4096
#define E_   8
#define SUBD 128
#define SUBH 512
#define M_   (B_ * S_)   // 16384 rows
#define SD_  (S_ * D_)   // 2097152

__device__ __forceinline__ unsigned short f2bf(float f) {
    unsigned int u = __float_as_uint(f);
    unsigned int r = u + 0x7FFFu + ((u >> 16) & 1u);   // round-to-nearest-even
    return (unsigned short)(r >> 16);
}

// gfx1250 hardware transcendental tanh (TRANS32). v_nop covers the
// trans-result hazard (ISA: 1 independent op or V_NOP before output use).
__device__ __forceinline__ float tanh_hw(float x) {
    float r;
    asm volatile("v_tanh_f32 %0, %1\n\tv_nop" : "=v"(r) : "v"(x));
    return r;
}

// tanh-form GELU: branchless, 1 TRANS + ~8 VALU (vs ~40 branchy ops for erff)
__device__ __forceinline__ float gelu_f(float v) {
    float u = 0.7978845608028654f * (v + 0.044715f * v * v * v);
    return 0.5f * v * (1.0f + tanh_hw(u));
}

union FragAB {
    v16bf bf;
    uint4 q[2];
};

// Load a 16x32 bf16 WMMA fragment from row-major storage (ld in elements).
// A-matrix: idx axis = M rows (lane%16); lanes 0-15 hold K=k0+{0..7} and
// k0+16+{0..7}; lanes 16-31 hold K=k0+8+{0..7} and k0+24+{0..7}.
// B-matrix: same pattern with idx axis = N columns from [N,K]-transposed storage.
__device__ __forceinline__ void ld_frag(FragAB& f, const unsigned short* base,
                                        int ld, int idx0, int k0) {
    const int lane = threadIdx.x & 31;
    const int i  = idx0 + (lane & 15);
    const int kb = k0 + ((lane >> 4) << 3);
    const unsigned short* p = base + (size_t)i * ld + kb;
    f.q[0] = *reinterpret_cast<const uint4*>(p);
    f.q[1] = *reinterpret_cast<const uint4*>(p + 16);
}

__device__ __forceinline__ v8f wmma_bf16(const FragAB& a, const FragAB& b, v8f c) {
    return __builtin_amdgcn_wmma_f32_16x16x32_bf16(false, a.bf, false, b.bf,
                                                   (short)0, c, false, false);
}

// ---------------- routing ----------------

__global__ void k_zero(float* p, int n) {
    int i = blockIdx.x * blockDim.x + threadIdx.x;
    if (i < n) p[i] = 0.0f;
}

__global__ void k_route(const float* __restrict__ x, const float* __restrict__ wsw,
                        float* __restrict__ logits) {
    float acc[B_ * E_];
#pragma unroll
    for (int j = 0; j < B_ * E_; ++j) acc[j] = 0.0f;
    const int stride = gridDim.x * blockDim.x;
    for (int i = blockIdx.x * blockDim.x + threadIdx.x; i < SD_; i += stride) {
        float4 w0 = *reinterpret_cast<const float4*>(wsw + (size_t)i * E_);
        float4 w1 = *reinterpret_cast<const float4*>(wsw + (size_t)i * E_ + 4);
#pragma unroll
        for (int b = 0; b < B_; ++b) {
            float xv = x[(size_t)b * SD_ + i];
            acc[b * E_ + 0] += xv * w0.x; acc[b * E_ + 1] += xv * w0.y;
            acc[b * E_ + 2] += xv * w0.z; acc[b * E_ + 3] += xv * w0.w;
            acc[b * E_ + 4] += xv * w1.x; acc[b * E_ + 5] += xv * w1.y;
            acc[b * E_ + 6] += xv * w1.z; acc[b * E_ + 7] += xv * w1.w;
        }
    }
#pragma unroll
    for (int j = 0; j < B_ * E_; ++j) {
        float v = acc[j];
        for (int off = 16; off > 0; off >>= 1) v += __shfl_down(v, off, 32);
        if ((threadIdx.x & 31) == 0) atomicAdd(&logits[j], v);
    }
}

// softmax is monotonic -> top-k of logits == top-k of softmax; only the mask matters.
__global__ void k_mask(const float* __restrict__ logits, const float* __restrict__ bsw,
                       const int* __restrict__ topk, float* __restrict__ mask) {
    int b = threadIdx.x;
    if (b >= B_) return;
    float l[E_];
    bool used[E_];
#pragma unroll
    for (int e = 0; e < E_; ++e) {
        l[e] = logits[b * E_ + e] + bsw[e];
        used[e] = false;
        mask[b * E_ + e] = 0.0f;
    }
    int K = topk[0];
    for (int kk = 0; kk < K; ++kk) {
        float best = -3.4e38f;
        int bi = -1;
#pragma unroll
        for (int e = 0; e < E_; ++e)
            if (!used[e] && l[e] > best) { best = l[e]; bi = e; }   // strict '>' keeps lowest index on ties
        if (bi >= 0) { used[bi] = true; mask[b * E_ + bi] = 1.0f; }
    }
}

// ---------------- bf16 conversions ----------------

__global__ void k_conv_x4(const float* __restrict__ x, unsigned short* __restrict__ o, int n4) {
    int i = blockIdx.x * blockDim.x + threadIdx.x;
    if (i >= n4) return;
    float4 v = reinterpret_cast<const float4*>(x)[i];
    unsigned int a = (unsigned int)f2bf(v.x) | ((unsigned int)f2bf(v.y) << 16);
    unsigned int b = (unsigned int)f2bf(v.z) | ((unsigned int)f2bf(v.w) << 16);
    reinterpret_cast<uint2*>(o)[i] = make_uint2(a, b);
}

__global__ void k_conv_w1(const float* __restrict__ w1, unsigned short* __restrict__ w1T) {
    int i = blockIdx.x * blockDim.x + threadIdx.x;           // [E][128][512] -> [E][512][128]
    if (i >= E_ * SUBD * SUBH) return;
    int e = i / (SUBD * SUBH), rem = i % (SUBD * SUBH);
    int d = rem / SUBH, h = rem % SUBH;
    w1T[(size_t)e * SUBH * SUBD + (size_t)h * SUBD + d] = f2bf(w1[i]);
}

__global__ void k_conv_w2(const float* __restrict__ w2, unsigned short* __restrict__ w2T) {
    int i = blockIdx.x * blockDim.x + threadIdx.x;           // [E][512][512] (h,g) -> (g,h)
    if (i >= E_ * SUBH * SUBH) return;
    int e = i / (SUBH * SUBH), rem = i % (SUBH * SUBH);
    int h = rem / SUBH, g = rem % SUBH;
    w2T[(size_t)e * SUBH * SUBH + (size_t)g * SUBH + h] = f2bf(w2[i]);
}

__global__ void k_conv_wout(const float* __restrict__ wo, unsigned short* __restrict__ woT) {
    int i = blockIdx.x * blockDim.x + threadIdx.x;           // [H][D] -> [D][H]
    if (i >= H_ * D_) return;
    int k = i / D_, n = i % D_;
    woT[(size_t)n * H_ + k] = f2bf(wo[i]);
}

// ---------------- fused expert MLP (per expert, masked) ----------------

__global__ __launch_bounds__(128, 1)
void k_expert(const unsigned short* __restrict__ xbf, const unsigned short* __restrict__ w1T,
              const unsigned short* __restrict__ w2T, const float* __restrict__ b1,
              const float* __restrict__ b2, const float* __restrict__ mask,
              unsigned short* __restrict__ hbuf) {
    __shared__ unsigned short h1s[4][16 * SUBH];   // 64 KB: per-wave 16x512 bf16 staging
    const int e    = blockIdx.y;
    const int row0 = blockIdx.x * 64;
    const int b    = row0 / S_;
    const int w    = threadIdx.x >> 5;
    const int lane = threadIdx.x & 31;

    if (mask[b * E_ + e] == 0.0f) {
        // masked expert: gelu(0)=0 -> write zeros, skip all compute (exact)
        uint4 z = make_uint4(0u, 0u, 0u, 0u);
        for (int i = threadIdx.x; i < (64 * SUBH) / 8; i += blockDim.x) {
            int flat = i * 8, r = flat >> 9, c = flat & 511;
            *reinterpret_cast<uint4*>(hbuf + (size_t)(row0 + r) * H_ + e * SUBH + c) = z;
        }
        return;
    }

    const int rbase = row0 + w * 16;
    const int n_lo  = lane & 15;
    const int hi    = lane >> 4;

    // A fragments of X slice: 16 rows x 128 K (held in VGPRs for all of GEMM1)
    FragAB ax[4];
#pragma unroll
    for (int kt = 0; kt < 4; ++kt)
        ld_frag(ax[kt], xbf + (size_t)rbase * D_ + e * SUBD, D_, 0, kt * 32);

    unsigned short* myh1 = &h1s[w][0];
    const unsigned short* w1e = w1T + (size_t)e * SUBH * SUBD;
    const float* b1e = b1 + e * SUBH;

    // GEMM1: h1 = gelu(X @ w1 + b1), stored bf16 in LDS
#pragma unroll 1
    for (int nt = 0; nt < SUBH / 16; ++nt) {
        v8f acc = {0.f, 0.f, 0.f, 0.f, 0.f, 0.f, 0.f, 0.f};
#pragma unroll
        for (int kt = 0; kt < 4; ++kt) {
            FragAB bf_;
            ld_frag(bf_, w1e, SUBD, nt * 16, kt * 32);
            acc = wmma_bf16(ax[kt], bf_, acc);
        }
        const int n = nt * 16 + n_lo;
        const float bias = b1e[n];
#pragma unroll
        for (int r = 0; r < 8; ++r)
            myh1[(size_t)(r + hi * 8) * SUBH + n] = f2bf(gelu_f(acc[r] + bias));
    }

    const unsigned short* w2e = w2T + (size_t)e * SUBH * SUBH;
    const float* b2e = b2 + e * SUBH;

    // GEMM2: h2 = h1 @ w2 + b2 ; hbuf = gelu(h2)  (mask==1 here)
    // Two N-tiles per pass: one A fragment feeds two consecutive WMMAs.
#pragma unroll 1
    for (int nt = 0; nt < SUBH / 16; nt += 2) {
        v8f acc0 = {0.f, 0.f, 0.f, 0.f, 0.f, 0.f, 0.f, 0.f};
        v8f acc1 = {0.f, 0.f, 0.f, 0.f, 0.f, 0.f, 0.f, 0.f};
#pragma unroll 2
        for (int kt = 0; kt < SUBH / 32; ++kt) {
            FragAB a2, b0, b1f;
            ld_frag(a2,  myh1, SUBH, 0, kt * 32);
            ld_frag(b0,  w2e,  SUBH, nt * 16,       kt * 32);
            ld_frag(b1f, w2e,  SUBH, (nt + 1) * 16, kt * 32);
            acc0 = wmma_bf16(a2, b0,  acc0);
            acc1 = wmma_bf16(a2, b1f, acc1);
        }
        const int n0 = nt * 16 + n_lo;
        const int n1 = n0 + 16;
        const float bias0 = b2e[n0];
        const float bias1 = b2e[n1];
#pragma unroll
        for (int r = 0; r < 8; ++r) {
            size_t rowoff = (size_t)(rbase + r + hi * 8) * H_ + e * SUBH;
            hbuf[rowoff + n0] = f2bf(gelu_f(acc0[r] + bias0));
            hbuf[rowoff + n1] = f2bf(gelu_f(acc1[r] + bias1));
        }
    }
}

// ---------------- output GEMM: out = gelu(h) @ w_out + b_out ----------------

__global__ __launch_bounds__(256, 1)
void k_out(const unsigned short* __restrict__ hbuf, const unsigned short* __restrict__ woT,
           const float* __restrict__ bout, float* __restrict__ out) {
    const int w    = threadIdx.x >> 5;
    const int lane = threadIdx.x & 31;
    const int m0   = blockIdx.y * 32  + (w & 1) * 16;
    const int n0   = blockIdx.x * 128 + (w >> 1) * 32;   // two 16-wide N tiles per wave

    v8f acc0 = {0.f, 0.f, 0.f, 0.f, 0.f, 0.f, 0.f, 0.f};
    v8f acc1 = {0.f, 0.f, 0.f, 0.f, 0.f, 0.f, 0.f, 0.f};
#pragma unroll 2
    for (int kt = 0; kt < H_ / 32; ++kt) {
        FragAB a, b0, b1;
        ld_frag(a,  hbuf + (size_t)m0 * H_,        H_, 0, kt * 32);
        ld_frag(b0, woT  + (size_t)n0 * H_,        H_, 0, kt * 32);
        ld_frag(b1, woT  + (size_t)(n0 + 16) * H_, H_, 0, kt * 32);
        acc0 = wmma_bf16(a, b0, acc0);
        acc1 = wmma_bf16(a, b1, acc1);
    }
    const int na = n0 + (lane & 15);
    const int nb = na + 16;
    const int hi = lane >> 4;
    const float biasa = bout[na];
    const float biasb = bout[nb];
#pragma unroll
    for (int r = 0; r < 8; ++r) {
        size_t rowoff = (size_t)(m0 + r + hi * 8) * D_;
        out[rowoff + na] = acc0[r] + biasa;
        out[rowoff + nb] = acc1[r] + biasb;
    }
}

// ---------------- launch ----------------

extern "C" void kernel_launch(void* const* d_in, const int* in_sizes, int n_in,
                              void* d_out, int out_size, void* d_ws, size_t ws_size,
                              hipStream_t stream) {
    const float* x    = (const float*)d_in[0];
    const float* wsw  = (const float*)d_in[1];
    const float* bsw  = (const float*)d_in[2];
    const float* w1   = (const float*)d_in[3];
    const float* b1   = (const float*)d_in[4];
    const float* w2   = (const float*)d_in[5];
    const float* b2   = (const float*)d_in[6];
    const float* wout = (const float*)d_in[7];
    const float* bout = (const float*)d_in[8];
    const int*   topk = (const int*)d_in[9];
    float* out = (float*)d_out;

    char* ws = (char*)d_ws;
    size_t off = 0;
    auto alloc = [&](size_t bytes) -> void* {
        void* p = ws + off;
        off += (bytes + 255) & ~(size_t)255;
        return p;
    };
    float* logits = (float*)alloc(B_ * E_ * sizeof(float));
    float* maskb  = (float*)alloc(B_ * E_ * sizeof(float));
    unsigned short* xbf  = (unsigned short*)alloc((size_t)M_ * D_ * 2);
    unsigned short* w1T  = (unsigned short*)alloc((size_t)E_ * SUBH * SUBD * 2);
    unsigned short* w2T  = (unsigned short*)alloc((size_t)E_ * SUBH * SUBH * 2);
    unsigned short* woT  = (unsigned short*)alloc((size_t)D_ * H_ * 2);
    unsigned short* hbuf = (unsigned short*)alloc((size_t)M_ * H_ * 2);

    k_zero<<<1, 64, 0, stream>>>(logits, B_ * E_);
    k_route<<<512, 256, 0, stream>>>(x, wsw, logits);
    k_mask<<<1, 32, 0, stream>>>(logits, bsw, topk, maskb);

    k_conv_x4  <<<(M_ * D_ / 4 + 255) / 256, 256, 0, stream>>>(x, xbf, M_ * D_ / 4);
    k_conv_w1  <<<(E_ * SUBD * SUBH + 255) / 256, 256, 0, stream>>>(w1, w1T);
    k_conv_w2  <<<(E_ * SUBH * SUBH + 255) / 256, 256, 0, stream>>>(w2, w2T);
    k_conv_wout<<<(H_ * D_ + 255) / 256, 256, 0, stream>>>(wout, woT);

    dim3 ge(M_ / 64, E_);
    k_expert<<<ge, 128, 0, stream>>>(xbf, w1T, w2T, b1, b2, maskb, hbuf);

    dim3 go(D_ / 128, M_ / 32);
    k_out<<<go, 256, 0, stream>>>(hbuf, woT, bout, out);
}
